// BidirRWKV6GaussianTimeMix_91268055040693
// MI455X (gfx1250) — matrix operand order repaired and checked by reading, not compiled, not run on hardware
//
#include <hip/hip_runtime.h>
#include <hip/hip_bf16.h>

// ---------------------------------------------------------------------------
// Problem constants (match reference)
// ---------------------------------------------------------------------------
#define Bb 4
#define Tt 1024
#define Dd 1024
#define Hh 16
#define Kk 64
#define DMIX 32
#define N5 160          // DMIX*5
#define DDEC 64
#define MID 512         // T/2
#define Mrows (Bb*Tt)   // 4096
#define EPSV 6.4e-4f    // 1e-5 * 8^2

typedef __attribute__((ext_vector_type(16))) _Float16 v16h;
typedef __attribute__((ext_vector_type(8)))  _Float16 v8h;
typedef __attribute__((ext_vector_type(2)))  _Float16 v2h;
typedef __attribute__((ext_vector_type(8)))  float    v8f;
typedef __attribute__((ext_vector_type(2)))  float    v2f;

__device__ inline v8f wmma_f16(v16h a, v16h b, v8f c) {
    return __builtin_amdgcn_wmma_f32_16x16x32_f16(false, a, false, b, (short)0, c,
                                                  false, false);
}

// ---------------------------------------------------------------------------
// CDNA5 async global->LDS copy (ASYNCcnt-tracked DMA path, ISA 15.18.3)
// vdst = per-lane LDS byte address; vaddr = per-lane 64-bit global address.
// ---------------------------------------------------------------------------
__device__ inline void async_copy_b128(const void* gsrc, void* lds_dst) {
    unsigned lds = (unsigned)(size_t)lds_dst;   // wave-relative LDS address
    asm volatile("global_load_async_to_lds_b128 %0, %1, off"
                 :
                 : "v"(lds), "v"(gsrc)
                 : "memory");
}
__device__ inline void wait_asynccnt0() {
    asm volatile("s_wait_asynccnt 0x0" ::: "memory");
}

// ---------------------------------------------------------------------------
// Fragment loaders (contiguous k = 16*hi + s per lane, wave32)
// ---------------------------------------------------------------------------
// A fragment from a row-major (rows x ld) f16 region at column offset koff:
// lane (m = lane&15, hi = lane>>4) takes halves [koff+hi*16, +16)
// -> two aligned 16B loads.
__device__ inline v16h load_a_frag(const _Float16* p, int ld, int koff, int lane) {
    int m = lane & 15, hi = lane >> 4;
    const _Float16* row = p + (size_t)m * ld + koff + hi * 16;
    v8h a0 = *(const v8h*)row;
    v8h a1 = *(const v8h*)(row + 8);
    return __builtin_shufflevector(a0, a1, 0, 1, 2, 3, 4, 5, 6, 7, 8, 9, 10, 11,
                                   12, 13, 14, 15);
}

// B fragment of Xᵀ where X is row-major (16 rows x ld): B[k=c][n=row]
// lane (n = lane&15, hi): halves [koff+hi*16, +16) of row n -> 2x16B loads.
__device__ inline v16h load_bT_frag(const _Float16* p, int ld, int koff,
                                    int lane) {
    int n = lane & 15, hi = lane >> 4;
    const _Float16* row = p + (size_t)n * ld + koff + hi * 16;
    v8h b0 = *(const v8h*)row;
    v8h b1 = *(const v8h*)(row + 8);
    return __builtin_shufflevector(b0, b1, 0, 1, 2, 3, 4, 5, 6, 7, 8, 9, 10, 11,
                                   12, 13, 14, 15);
}

// B fragment from channel-major storage X[c][t] (ld = row length in tokens):
// B[k = token][n = channel]; lane n picks channel row, elements run along t.
__device__ inline v16h load_bCM_frag(const _Float16* p, int ld, int toff,
                                     int lane) {
    int n = lane & 15, hi = lane >> 4;
    const _Float16* row = p + (size_t)n * ld + toff + hi * 16;
    v8h b0 = *(const v8h*)row;
    v8h b1 = *(const v8h*)(row + 8);
    return __builtin_shufflevector(b0, b1, 0, 1, 2, 3, 4, 5, 6, 7, 8, 9, 10, 11,
                                   12, 13, 14, 15);
}

// ---------------------------------------------------------------------------
// Tiled WMMA GEMM:  C[M,N] = act(A[M,K] @ B[K,N]),  B passed TRANSPOSED (N x K).
// Block: 256 threads = 8 waves; block tile 64(M) x 128(N); K stepped by 32.
// A tile staged via async global->LDS DMA (ASYNCcnt); B staged via VGPRs.
// ACT: 0=none(f32), 1=tanh(f32), 2=silu(f32), 3=mix-combine(f16):
//      C16 = f16( x + dxprev * (maa[col] + acc) )
// ---------------------------------------------------------------------------
template <int ACT>
__global__ __launch_bounds__(256) void gemm_wmma(
    const _Float16* __restrict__ A, int lda, const _Float16* __restrict__ Bt,
    int ldb, float* __restrict__ C, _Float16* __restrict__ C16, int ldc, int M,
    int N, int Kd, const float* __restrict__ xres,
    const float* __restrict__ dxprev, const float* __restrict__ maa) {
    __shared__ _Float16 sA[64 * 32];    // row-major, ld = 32
    __shared__ _Float16 sBt[128 * 32];  // [n][k], ld = 32
    const int tid = threadIdx.x;
    const int lane = tid & 31;
    const int wave = tid >> 5;
    const int row0 = blockIdx.y * 64;
    const int col0 = blockIdx.x * 128;

    // per-thread staging coordinates
    const int ar = (tid * 8) >> 5;         // A row in tile (0..63)
    const int ak = (tid * 8) & 31;         // A col in tile (0,8,16,24)
    const int bn = tid >> 1;               // B column (0..127)
    const int bk = (tid & 1) * 16;         // B k-offset (0 or 16)

    v8f acc[4] = {{}, {}, {}, {}};
    for (int k0 = 0; k0 < Kd; k0 += 32) {
        // stage A 64x32: async DMA straight into LDS (16B per thread)
        async_copy_b128(A + (size_t)(row0 + ar) * lda + k0 + ak, sA + tid * 8);
        // stage Bt 128x32 (two aligned 16B copies per thread), guard N
        {
            v8h lo = {}, hiv = {};
            int gcol = col0 + bn;
            if (gcol < N) {
                const _Float16* src = Bt + (size_t)gcol * ldb + k0 + bk;
                lo = *(const v8h*)src;
                hiv = *(const v8h*)(src + 8);
            }
            *(v8h*)(sBt + bn * 32 + bk) = lo;
            *(v8h*)(sBt + bn * 32 + bk + 8) = hiv;
        }
        // prefetch next K-step sources into cache
        if (k0 + 32 < Kd) {
            __builtin_prefetch(A + (size_t)(row0 + ar) * lda + k0 + 32 + ak, 0, 1);
            if (col0 + bn < N)
                __builtin_prefetch(Bt + (size_t)(col0 + bn) * ldb + k0 + 32 + bk,
                                   0, 1);
        }
        wait_asynccnt0();   // our async writes to sA are visible
        __syncthreads();    // all waves' staging complete

        // B fragment: 16 contiguous halves (two 16B LDS loads)
        v16h bf;
        {
            int n = lane & 15, hi = lane >> 4;
            const _Float16* p = sBt + (wave * 16 + n) * 32 + hi * 16;
            v8h b0 = *(const v8h*)p;
            v8h b1 = *(const v8h*)(p + 8);
            bf = __builtin_shufflevector(b0, b1, 0, 1, 2, 3, 4, 5, 6, 7, 8, 9, 10,
                                         11, 12, 13, 14, 15);
        }
#pragma unroll
        for (int s = 0; s < 4; ++s) {
            v16h af = load_a_frag(sA + s * 16 * 32, 32, 0, lane);
            acc[s] = wmma_f16(af, bf, acc[s]);
        }
        __syncthreads();
    }

    // epilogue
    int n = lane & 15, hi = lane >> 4;
    int gc = col0 + wave * 16 + n;
    if (gc < N) {
#pragma unroll
        for (int s = 0; s < 4; ++s) {
#pragma unroll
            for (int vi = 0; vi < 8; ++vi) {
                int gr = row0 + s * 16 + vi + hi * 8;
                float v = acc[s][vi];
                if constexpr (ACT == 1) v = tanhf(v);
                if constexpr (ACT == 2) v = v / (1.f + __expf(-v));
                if constexpr (ACT == 3) {
                    size_t idx = (size_t)gr * ldc + gc;
                    float xv = xres[idx], dxv = dxprev[idx];
                    C16[idx] = (_Float16)(xv + dxv * (maa[gc] + v));
                } else {
                    C[(size_t)gr * ldc + gc] = v;
                }
            }
        }
    }
}

// ---------------------------------------------------------------------------
// Elementwise / utility kernels
// ---------------------------------------------------------------------------
__global__ void cvt_f32_f16(const float* __restrict__ in,
                            _Float16* __restrict__ out, int n) {
    int i = blockIdx.x * 256 + threadIdx.x;
    if (i < n) out[i] = (_Float16)in[i];
}

// in: (Kd x N) row-major f32 -> out: (N x Kd) row-major f16 (transposed copy)
__global__ void cvt_transpose_f32_f16(const float* __restrict__ in,
                                      _Float16* __restrict__ out, int Kd,
                                      int N) {
    int i = blockIdx.x * 256 + threadIdx.x;
    if (i >= Kd * N) return;
    int n = i / Kd, k = i % Kd;
    out[i] = (_Float16)in[(size_t)k * N + n];
}

// bidirectional token shift: dxprev = 0.5*(x_prev + x_next) - x ; xmx16 = f16(x + dxprev*maa_x)
__global__ void prep_kernel(const float* __restrict__ x,
                            const float* __restrict__ maa_x,
                            float* __restrict__ dxprev,
                            _Float16* __restrict__ xmx16) {
    size_t i = (size_t)blockIdx.x * 256 + threadIdx.x;
    if (i >= (size_t)Bb * Tt * Dd) return;
    int d = (int)(i % Dd);
    size_t bt = i / Dd;
    int t = (int)(bt % Tt);
    float xf = (t > 0) ? x[i - Dd] : 0.f;
    float xb = (t < Tt - 1) ? x[i + Dd] : 0.f;
    float dx = 0.5f * (xf + xb) - x[i];
    dxprev[i] = dx;
    xmx16[i] = (_Float16)(x[i] + dx * maa_x[d]);
}

// per-(b,h) sequential cumsum of w = -exp(decay[d] + wraw)
__global__ __launch_bounds__(64) void scan_kernel(
    const float* __restrict__ wraw, const float* __restrict__ tdecay,
    float* __restrict__ cs, float* __restrict__ csb) {
    int bh = blockIdx.x;          // b*H + h
    int b = bh >> 4, h = bh & 15;
    int c = threadIdx.x;
    int d = h * Kk + c;
    float run = 0.f;
    for (int t = 0; t < Tt; ++t) {
        float wv = -__expf(tdecay[d] + wraw[((size_t)(b * Tt + t)) * Dd + d]);
        float prev = run;
        run += wv;
        size_t o = (((size_t)bh) * Tt + t) * Kk + c;
        cs[o] = run;
        csb[o] = prev;
    }
}

__device__ inline float clamp60(float x) {
    return fminf(fmaxf(x, -60.f), 60.f);
}

// build decay-scaled r/k (fwd+bwd) in (B,H,T,K) and V in (B,H,K,T) f16 layout
__global__ void scale_kernel(const float* __restrict__ r,
                             const float* __restrict__ k,
                             const float* __restrict__ v,
                             const float* __restrict__ cs,
                             const float* __restrict__ csb,
                             _Float16* __restrict__ rf, _Float16* __restrict__ kf,
                             _Float16* __restrict__ rb, _Float16* __restrict__ kb,
                             _Float16* __restrict__ vT) {
    size_t i = (size_t)blockIdx.x * 256 + threadIdx.x;
    if (i >= (size_t)Bb * Hh * Tt * Kk) return;
    int c = (int)(i & (Kk - 1));
    size_t tmp = i >> 6;
    int t = (int)(tmp % Tt);
    int bh = (int)(tmp / Tt);
    int b = bh >> 4, h = bh & 15;
    size_t btd = ((size_t)(b * Tt + t)) * Dd + h * Kk + c;
    size_t midx = (((size_t)bh) * Tt + MID) * Kk + c;
    float fs = clamp60(cs[i] - cs[midx]);
    float bs = clamp60(csb[i] - csb[midx]);
    float rv = r[btd], kv = k[btd];
    rf[i] = (_Float16)(rv * __expf(fs));
    kf[i] = (_Float16)(kv * __expf(-fs));
    rb[i] = (_Float16)(rv * __expf(-bs));
    kb[i] = (_Float16)(kv * __expf(bs));
    // channel-major V for the AV stage: vT[(bh*K + c)*T + t]
    vT[((size_t)bh * Kk + c) * Tt + t] = (_Float16)v[btd];
}

// ---------------------------------------------------------------------------
// Attention: one wave per (b, h, i-tile).  A = sel(R̃K̃ᵀ) * gauss;  Y += A V
// ---------------------------------------------------------------------------
__global__ __launch_bounds__(32) void attn_kernel(
    const _Float16* __restrict__ rf, const _Float16* __restrict__ kf,
    const _Float16* __restrict__ rb, const _Float16* __restrict__ kb,
    const _Float16* __restrict__ vT, const float* __restrict__ mu_raw,
    const float* __restrict__ sig_raw, float* __restrict__ Y) {
    __shared__ _Float16 sA[16 * 32];
    int blk = blockIdx.x;            // bh*64 + itile
    int it = blk & 63;
    int bh = blk >> 6;
    int h = bh & (Hh - 1);
    int lane = threadIdx.x;
    int n = lane & 15, hi = lane >> 4;
    const size_t base = (size_t)bh * Tt * Kk;   // == bh*K*T as well

    float mu = logf(1.f + __expf(mu_raw[h]));
    float sg = logf(1.f + __expf(sig_raw[h]));
    float inv_sg = 1.f / sg;

    // i-tile row fragments (K = 64 -> two 32-wide k-steps)
    const _Float16* rfp = rf + base + (size_t)it * 16 * Kk;
    const _Float16* rbp = rb + base + (size_t)it * 16 * Kk;
    v16h arf0 = load_a_frag(rfp, Kk, 0, lane);
    v16h arf1 = load_a_frag(rfp, Kk, 32, lane);
    v16h arb0 = load_a_frag(rbp, Kk, 0, lane);
    v16h arb1 = load_a_frag(rbp, Kk, 32, lane);

    v8f accY[4] = {{}, {}, {}, {}};

    for (int jp = 0; jp < 32; ++jp) {
#pragma unroll
        for (int jj = 0; jj < 2; ++jj) {
            int j = jp * 2 + jj;
            v8f aF = {};
            v8f aB = {};
            if (j <= it) {
                const _Float16* kp = kf + base + (size_t)j * 16 * Kk;
                aF = wmma_f16(arf0, load_bT_frag(kp, Kk, 0, lane), aF);
                aF = wmma_f16(arf1, load_bT_frag(kp, Kk, 32, lane), aF);
            }
            if (j >= it) {
                const _Float16* kp = kb + base + (size_t)j * 16 * Kk;
                aB = wmma_f16(arb0, load_bT_frag(kp, Kk, 0, lane), aB);
                aB = wmma_f16(arb1, load_bT_frag(kp, Kk, 32, lane), aB);
            }
            bool diag = (j == it);
#pragma unroll
            for (int vi = 0; vi < 8; ++vi) {
                int row = it * 16 + vi + hi * 8;
                int col = j * 16 + n;
                float a;
                if (diag)
                    a = (row >= col) ? aF[vi] : aB[vi];
                else
                    a = (j < it) ? aF[vi] : aB[vi];
                float dist = fabsf((float)(row - col));
                float z = (dist - mu) * inv_sg;
                a *= __expf(-0.5f * z * z);
                sA[(vi + hi * 8) * 32 + jj * 16 + n] = (_Float16)a;
            }
        }
        __syncthreads();
        v16h af = load_a_frag(sA, 32, 0, lane);
        int jrow0 = jp * 32;
#pragma unroll
        for (int ct = 0; ct < 4; ++ct) {
            // vT channel-major: contiguous along tokens -> 2x16B loads
            v16h bf = load_bCM_frag(vT + base + (size_t)ct * 16 * Tt, Tt, jrow0,
                                    lane);
            accY[ct] = wmma_f16(af, bf, accY[ct]);
        }
        __syncthreads();
    }

    // store Y tile (16 x 64)
#pragma unroll
    for (int ct = 0; ct < 4; ++ct)
#pragma unroll
        for (int vi = 0; vi < 8; ++vi)
            Y[base + (size_t)(it * 16 + vi + hi * 8) * Kk + ct * 16 + n] =
                accY[ct][vi];
}

// ---------------------------------------------------------------------------
// GroupNorm over K=64 per head (wave32 shuffle reduction, 2 channels/lane),
// scale/shift, multiply by silu gate, -> f16.  One wave per (b,t,h) group.
// ---------------------------------------------------------------------------
__global__ __launch_bounds__(256) void gnorm_kernel(
    const float* __restrict__ Y, const float* __restrict__ gsil,
    const float* __restrict__ lnw, const float* __restrict__ lnb,
    _Float16* __restrict__ yg) {
    int gid = blockIdx.x * 8 + (threadIdx.x >> 5);  // (b*T + t)*H + h
    int lane = threadIdx.x & 31;
    int h = gid % Hh;
    size_t bt = gid / Hh;
    int t = (int)(bt % Tt);
    int b = (int)(bt / Tt);

    size_t ybase = ((((size_t)b * Hh + h) * Tt) + t) * Kk + lane * 2;
    v2f y2 = *(const v2f*)(Y + ybase);

    float s = y2[0] + y2[1];
#pragma unroll
    for (int m = 16; m > 0; m >>= 1) s += __shfl_xor(s, m, 32);
    float mean = s * (1.f / 64.f);
    float d0 = y2[0] - mean, d1 = y2[1] - mean;
    float q = d0 * d0 + d1 * d1;
#pragma unroll
    for (int m = 16; m > 0; m >>= 1) q += __shfl_xor(q, m, 32);
    float inv = rsqrtf(q * (1.f / 64.f) + EPSV);

    int d = h * Kk + lane * 2;
    size_t gi = ((size_t)b * Tt + t) * Dd + d;
    v2f g2 = *(const v2f*)(gsil + gi);
    v2h o;
    o[0] = (_Float16)((d0 * inv * lnw[d] + lnb[d]) * g2[0]);
    o[1] = (_Float16)((d1 * inv * lnw[d + 1] + lnb[d + 1]) * g2[1]);
    *(v2h*)(yg + gi) = o;
}

// ---------------------------------------------------------------------------
// Host-side orchestration
// ---------------------------------------------------------------------------
static inline size_t alignup(size_t x) { return (x + 255) & ~(size_t)255; }

extern "C" void kernel_launch(void* const* d_in, const int* in_sizes, int n_in,
                              void* d_out, int out_size, void* d_ws,
                              size_t ws_size, hipStream_t stream) {
    (void)in_sizes; (void)n_in; (void)out_size; (void)ws_size;
    const float* x       = (const float*)d_in[0];
    const float* maa_x   = (const float*)d_in[1];
    const float* maa5[5] = {(const float*)d_in[2], (const float*)d_in[3],
                            (const float*)d_in[4], (const float*)d_in[5],
                            (const float*)d_in[6]};
    const float* maa_w1  = (const float*)d_in[7];
    const float* maa_w2  = (const float*)d_in[8];
    const float* tdecay  = (const float*)d_in[9];
    const float* dec_w1  = (const float*)d_in[10];
    const float* dec_w2  = (const float*)d_in[11];
    const float* Wf[5]   = {(const float*)d_in[12], (const float*)d_in[13],
                            (const float*)d_in[14], (const float*)d_in[15],
                            (const float*)d_in[16]};
    const float* mu_raw  = (const float*)d_in[17];
    const float* sig_raw = (const float*)d_in[18];
    const float* lnw     = (const float*)d_in[19];
    const float* lnb     = (const float*)d_in[20];
    float* out = (float*)d_out;

    // ---- workspace carving ----
    char* w = (char*)d_ws;
    size_t off = 0;
    auto carve = [&](size_t bytes) {
        void* p = w + off;
        off += alignup(bytes);
        return p;
    };
    const size_t NBTD = (size_t)Bb * Tt * Dd;            // 4 Mi elements
    float*    dxprev = (float*)carve(NBTD * 4);
    _Float16* xmx16  = (_Float16*)carve(NBTD * 2);
    _Float16* Wt16[5];   // transposed (out x in) f16 weights
    for (int i = 0; i < 5; ++i) Wt16[i] = (_Float16*)carve((size_t)Dd * Dd * 2);
    _Float16* maaw1t = (_Float16*)carve((size_t)N5 * Dd * 2);       // 160x1024
    _Float16* maaw2t = (_Float16*)carve((size_t)5 * Dd * DMIX * 2); // 5x(1024x32)
    _Float16* decw1t = (_Float16*)carve((size_t)DDEC * Dd * 2);     // 64x1024
    _Float16* decw2t = (_Float16*)carve((size_t)Dd * DDEC * 2);     // 1024x64
    float*    xxx    = (float*)carve((size_t)Mrows * N5 * 4);
    _Float16* xxx16  = (_Float16*)carve((size_t)Mrows * N5 * 2);
    _Float16* x5[5];
    for (int i = 0; i < 5; ++i) x5[i] = (_Float16*)carve(NBTD * 2);
    float* rbuf = (float*)carve(NBTD * 4);
    float* kbuf = (float*)carve(NBTD * 4);
    float* vbuf = (float*)carve(NBTD * 4);
    float* gbuf = (float*)carve(NBTD * 4);
    float* h1   = (float*)carve((size_t)Mrows * DDEC * 4);
    _Float16* h1_16 = (_Float16*)carve((size_t)Mrows * DDEC * 2);
    float* wraw = (float*)carve(NBTD * 4);
    float* cs   = (float*)carve(NBTD * 4);
    float* csb  = (float*)carve(NBTD * 4);
    _Float16* rfb = (_Float16*)carve(NBTD * 2);
    _Float16* kfb = (_Float16*)carve(NBTD * 2);
    _Float16* rbb = (_Float16*)carve(NBTD * 2);
    _Float16* kbb = (_Float16*)carve(NBTD * 2);
    _Float16* vTb = (_Float16*)carve(NBTD * 2);
    // reuse: r/k f32 buffers are dead after scale_kernel
    float*    Ybuf = rbuf;
    _Float16* yg   = (_Float16*)kbuf;

    auto cdiv = [](int a, int b) { return (a + b - 1) / b; };

    // 1) weight conversions f32 -> f16 transposed (N x K)
    {
        int nW = Dd * Dd;
        for (int i = 0; i < 5; ++i)
            cvt_transpose_f32_f16<<<cdiv(nW, 256), 256, 0, stream>>>(Wf[i],
                                                                     Wt16[i], Dd,
                                                                     Dd);
        cvt_transpose_f32_f16<<<cdiv(Dd * N5, 256), 256, 0, stream>>>(
            maa_w1, maaw1t, Dd, N5);  // -> (160 x 1024)
        for (int f = 0; f < 5; ++f)
            cvt_transpose_f32_f16<<<cdiv(DMIX * Dd, 256), 256, 0, stream>>>(
                maa_w2 + (size_t)f * DMIX * Dd, maaw2t + (size_t)f * Dd * DMIX,
                DMIX, Dd);  // -> (1024 x 32)
        cvt_transpose_f32_f16<<<cdiv(Dd * DDEC, 256), 256, 0, stream>>>(
            dec_w1, decw1t, Dd, DDEC);  // -> (64 x 1024)
        cvt_transpose_f32_f16<<<cdiv(DDEC * Dd, 256), 256, 0, stream>>>(
            dec_w2, decw2t, DDEC, Dd);  // -> (1024 x 64)
    }

    // 2) token shift
    prep_kernel<<<cdiv((int)NBTD, 256), 256, 0, stream>>>(x, maa_x, dxprev,
                                                          xmx16);

    // 3) xxx = tanh(xmix @ maa_w1)   (4096 x 1024) @ (1024 x 160)
    {
        dim3 g(cdiv(N5, 128), Mrows / 64);
        gemm_wmma<1><<<g, 256, 0, stream>>>(xmx16, Dd, maaw1t, Dd, xxx, nullptr,
                                            N5, Mrows, N5, Dd, nullptr, nullptr,
                                            nullptr);
    }
    cvt_f32_f16<<<cdiv(Mrows * N5, 256), 256, 0, stream>>>(xxx, xxx16,
                                                           Mrows * N5);

    // 4) five mix GEMMs (K=32) fused with x + dxprev*(maa + m) -> f16
    for (int f = 0; f < 5; ++f) {
        dim3 g(cdiv(Dd, 128), Mrows / 64);
        gemm_wmma<3><<<g, 256, 0, stream>>>(
            xxx16 + f * DMIX, N5, maaw2t + (size_t)f * Dd * DMIX, DMIX, nullptr,
            x5[f], Dd, Mrows, Dd, DMIX, x, dxprev, maa5[f]);
    }
    // x5 order: 0=xw 1=xk 2=xv 3=xr 4=xg

    // 5) big projections r,k,v (plain) and g (silu)
    {
        dim3 g(cdiv(Dd, 128), Mrows / 64);
        gemm_wmma<0><<<g, 256, 0, stream>>>(x5[3], Dd, Wt16[0], Dd, rbuf, nullptr,
                                            Dd, Mrows, Dd, Dd, nullptr, nullptr,
                                            nullptr);
        gemm_wmma<0><<<g, 256, 0, stream>>>(x5[1], Dd, Wt16[1], Dd, kbuf, nullptr,
                                            Dd, Mrows, Dd, Dd, nullptr, nullptr,
                                            nullptr);
        gemm_wmma<0><<<g, 256, 0, stream>>>(x5[2], Dd, Wt16[2], Dd, vbuf, nullptr,
                                            Dd, Mrows, Dd, Dd, nullptr, nullptr,
                                            nullptr);
        gemm_wmma<2><<<g, 256, 0, stream>>>(x5[4], Dd, Wt16[3], Dd, gbuf, nullptr,
                                            Dd, Mrows, Dd, Dd, nullptr, nullptr,
                                            nullptr);
    }

    // 6) decay path: h1 = tanh(xw @ dec_w1); wraw = h1 @ dec_w2
    {
        dim3 g1(cdiv(DDEC, 128), Mrows / 64);
        gemm_wmma<1><<<g1, 256, 0, stream>>>(x5[0], Dd, decw1t, Dd, h1, nullptr,
                                             DDEC, Mrows, DDEC, Dd, nullptr,
                                             nullptr, nullptr);
        cvt_f32_f16<<<cdiv(Mrows * DDEC, 256), 256, 0, stream>>>(
            h1, h1_16, Mrows * DDEC);
        dim3 g2(cdiv(Dd, 128), Mrows / 64);
        gemm_wmma<0><<<g2, 256, 0, stream>>>(h1_16, DDEC, decw2t, DDEC, wraw,
                                             nullptr, Dd, Mrows, Dd, DDEC,
                                             nullptr, nullptr, nullptr);
    }

    // 7) per-(b,h) cumsum of -exp(decay + wraw)
    scan_kernel<<<Bb * Hh, 64, 0, stream>>>(wraw, tdecay, cs, csb);

    // 8) decay-scaled r/k (token-major) and V (channel-major) -> f16
    scale_kernel<<<cdiv((int)NBTD, 256), 256, 0, stream>>>(
        rbuf, kbuf, vbuf, cs, csb, rfb, kfb, rbb, kbb, vTb);

    // 9) attention (one wave per (b,h,i-tile))
    attn_kernel<<<Bb * Hh * (Tt / 16), 32, 0, stream>>>(rfb, kfb, rbb, kbb, vTb,
                                                        mu_raw, sig_raw, Ybuf);

    // 10) groupnorm + gate -> f16 (one wave per group, 8 groups/block)
    gnorm_kernel<<<(Bb * Tt * Hh) / 8, 256, 0, stream>>>(Ybuf, gbuf, lnw, lnb,
                                                         yg);

    // 11) output projection -> d_out (f32)
    {
        dim3 g(cdiv(Dd, 128), Mrows / 64);
        gemm_wmma<0><<<g, 256, 0, stream>>>(yg, Dd, Wt16[4], Dd, out, nullptr, Dd,
                                            Mrows, Dd, Dd, nullptr, nullptr,
                                            nullptr);
    }
}
